// GRUModel_57166014709914
// MI455X (gfx1250) — matrix operand order repaired and checked by reading, not compiled, not run on hardware
//
#include <hip/hip_runtime.h>

// ---------------------------------------------------------------------------
// Bidirectional 2-layer GRU + FC for MI455X (gfx1250).
//   - v_wmma_f32_16x16x32_bf16 for all GEMM work (f32 accumulate)
//   - per-workgroup activation tile staged in LDS via GLOBAL_LOAD_ASYNC_TO_LDS
//     (ASYNCcnt path), consumed with conflict-free ds_load_b128
//   - nontemporal stores for the ~872 MB of streamed outputs (protect L2)
// ---------------------------------------------------------------------------

typedef __bf16 bf16;
typedef bf16  v16bf __attribute__((ext_vector_type(16)));
typedef bf16  v8bf  __attribute__((ext_vector_type(8)));
typedef float v8f   __attribute__((ext_vector_type(8)));

#define DEVINL __device__ __forceinline__
#define USE_ASYNC_LDS 1

static constexpr int B_   = 128;
static constexpr int T_   = 512;
static constexpr int IN_  = 256;
static constexpr int H_   = 512;
static constexpr int OUT_ = 256;
static constexpr int H2_  = 2 * H_;

DEVINL v8f wmma_bf16(v16bf a, v16bf b, v8f c) {
  // (neg_a, A, neg_b, B, c_mod, C, reuse_a, reuse_b)
  return __builtin_amdgcn_wmma_f32_16x16x32_bf16(false, a, false, b, (short)0, c,
                                                 false, false);
}

// A operand: 16x32 (MxK) from row-major rows (global or LDS).
// ISA layout: lanes 0-15 = row lane,    K {k..k+7, k+16..k+23}
//             lanes 16-31 = row lane-16, K {k+8..k+15, k+24..k+31}
DEVINL v16bf load_a(const bf16* __restrict__ base, size_t ld, int lane, int k) {
  const int half = lane >> 4;
  const int r    = lane & 15;
  const bf16* p  = base + (size_t)r * ld + (size_t)(k + half * 8);
  v8bf lo = *(const v8bf*)(p);
  v8bf hi = *(const v8bf*)(p + 16);
  v16bf f;
#pragma unroll
  for (int i = 0; i < 8; ++i) { f[i] = lo[i]; f[8 + i] = hi[i]; }
  return f;
}

// B operand: 32x16 (KxN) from row-major (N,K) weights (B[k,n] = W[n,k]).
// lanes 0-15 = col n, K {k..k+7, k+16..k+23}; lanes 16-31 = col n, K {k+8..k+15, k+24..k+31}.
DEVINL v16bf load_b(const bf16* __restrict__ wbase, size_t ldk, int lane, int k) {
  const int half = lane >> 4;
  const int n    = lane & 15;
  const bf16* p  = wbase + (size_t)n * ldk + (size_t)(k + half * 8);
  v8bf lo = *(const v8bf*)(p);
  v8bf hi = *(const v8bf*)(p + 16);
  v16bf f;
#pragma unroll
  for (int i = 0; i < 8; ++i) { f[i] = lo[i]; f[8 + i] = hi[i]; }
  return f;
}

// Cooperative stage of a 16-row x cols tile into LDS (row pitch ldd elems).
// cols is a power of two multiple of 8; cshift = log2(cols/8).
DEVINL void stage16(bf16* __restrict__ dst, int ldd,
                    const bf16* __restrict__ src, size_t srow,
                    int cols, int cshift, int tid) {
  const int cpr   = cols >> 3;   // 16-byte chunks per row
  const int total = cpr << 4;    // 16 rows
  for (int c = tid; c < total; c += 256) {
    const int r = c >> cshift;
    const int q = c & (cpr - 1);
    const bf16* g = src + (size_t)r * srow + (size_t)(q << 3);
    bf16* l = dst + r * ldd + (q << 3);
#if USE_ASYNC_LDS
    // Memory -> LDS without VGPR bounce; tracked by ASYNCcnt.
    asm volatile("global_load_async_to_lds_b128 %0, %1, off"
                 :: "v"((unsigned)(uintptr_t)l), "v"(g) : "memory");
#else
    *(v8bf*)l = *(const v8bf*)g;
#endif
  }
}

DEVINL void stage_fence() {
#if USE_ASYNC_LDS
  asm volatile("s_wait_asynccnt 0x0" ::: "memory");
#endif
  __syncthreads();
}

DEVINL float sigmoidf_(float x) { return 1.0f / (1.0f + __expf(-x)); }

// ---------------------------------------------------------------------------
// One GRU timestep, one layer, both directions (blockIdx.z = dir).
// Fused gates = X_t @ Wih^T + H_prev @ Whh^T; n-gate input/hidden partials
// kept in separate accumulators (n = tanh(i_n + r*h_n)).
// Grid (8,4,2), block 256 (8 waves). Workgroup = 16 batch rows (staged in
// LDS), each wave owns 16 hidden columns across all 3 gates.
// ---------------------------------------------------------------------------
__global__ __launch_bounds__(256)
void gru_step(const bf16* __restrict__ Xin, int Kin, size_t xin_dir_stride,
              bf16* __restrict__ Hseq, size_t hseq_dir_stride, int ldh,
              int hcol_per_dir,
              const bf16* __restrict__ Wih,   // (3H, Kin) bf16 row-major
              const bf16* __restrict__ Whh,   // (3H, H)   bf16 row-major
              const float* __restrict__ bih, const float* __restrict__ bhh,
              int s, int write_outputs,
              float* __restrict__ gr, float* __restrict__ gz,
              float* __restrict__ hn) {
  constexpr int LDP = 520;               // 512 + 8 pad: conflict-free b128
  __shared__ bf16 sX[16 * LDP];          // phase-1 A tile (layer input)
  __shared__ bf16 sH[16 * LDP];          // phase-2 A tile (h_prev)

  const int dir = blockIdx.z;
  const int t   = dir ? (T_ - 1 - s) : s;
  const int tp  = dir ? (t + 1) : (t - 1);
  const bool first = (s == 0);

  const bf16* X  = Xin  + xin_dir_stride  * dir;
  bf16*       Hs = Hseq + hseq_dir_stride * dir;
  const int   hc = hcol_per_dir * dir;

  const int tid  = threadIdx.x;
  const int lane = tid & 31;
  const int w    = tid >> 5;
  const int M0   = blockIdx.x * 16;            // batch rows
  const int N0   = blockIdx.y * 128 + w * 16;  // hidden columns (wave tile)

  // ---- stage both activation tiles (async -> LDS), then fence ----
  const int csh1 = (Kin == 256) ? 5 : 6;
  stage16(sX, LDP, X + ((size_t)M0 * T_ + (size_t)t) * (size_t)Kin,
          (size_t)T_ * (size_t)Kin, Kin, csh1, tid);
  if (!first)
    stage16(sH, LDP, Hs + ((size_t)M0 * T_ + (size_t)tp) * (size_t)ldh + hc,
            (size_t)T_ * (size_t)ldh, H_, 6, tid);
  stage_fence();

  v8f aR, aZ, aNi, aNh;
  aR = 0.f; aZ = 0.f; aNi = 0.f; aNh = 0.f;

  // ---- phase 1: input contribution ----
  {
    const bf16* wR = Wih + (size_t)(0 * H_ + N0) * (size_t)Kin;
    const bf16* wZ = Wih + (size_t)(1 * H_ + N0) * (size_t)Kin;
    const bf16* wN = Wih + (size_t)(2 * H_ + N0) * (size_t)Kin;
    for (int k = 0; k < Kin; k += 32) {
      v16bf a = load_a(sX, LDP, lane, k);
      aR  = wmma_bf16(a, load_b(wR, Kin, lane, k), aR);
      aZ  = wmma_bf16(a, load_b(wZ, Kin, lane, k), aZ);
      aNi = wmma_bf16(a, load_b(wN, Kin, lane, k), aNi);
    }
  }

  // ---- phase 2: hidden contribution (skipped at s==0, h0 == 0) ----
  if (!first) {
    const bf16* uR = Whh + (size_t)(0 * H_ + N0) * (size_t)H_;
    const bf16* uZ = Whh + (size_t)(1 * H_ + N0) * (size_t)H_;
    const bf16* uN = Whh + (size_t)(2 * H_ + N0) * (size_t)H_;
    for (int k = 0; k < H_; k += 32) {
      v16bf a = load_a(sH, LDP, lane, k);
      aR  = wmma_bf16(a, load_b(uR, H_, lane, k), aR);
      aZ  = wmma_bf16(a, load_b(uZ, H_, lane, k), aZ);
      aNh = wmma_bf16(a, load_b(uN, H_, lane, k), aNh);
    }
  }

  // ---- epilogue: GRU cell on the 16x16 patch ----
  const int half = lane >> 4;
  const int col  = lane & 15;
  const int j    = N0 + col;
  const float bR  = bih[j]          + bhh[j];
  const float bZ  = bih[H_ + j]     + bhh[H_ + j];
  const float bNi = bih[2 * H_ + j];
  const float bNh = bhh[2 * H_ + j];
#pragma unroll
  for (int v = 0; v < 8; ++v) {
    const int m = half * 8 + v;
    const int b = M0 + m;
    const float rr = sigmoidf_(aR[v] + bR);
    const float zz = sigmoidf_(aZ[v] + bZ);
    const float nn = tanhf(aNi[v] + bNi + rr * (aNh[v] + bNh));
    const float hp = first ? 0.f : (float)sH[m * LDP + j];  // h_prev from LDS
    const float hy = nn + zz * (hp - nn);
    Hs[((size_t)b * T_ + (size_t)t) * (size_t)ldh + hc + j] = (bf16)hy;
    if (write_outputs) {
      const size_t o = ((size_t)b * T_ + (size_t)t) * (size_t)H2_ + hc + j;
      __builtin_nontemporal_store(rr, gr + o);
      __builtin_nontemporal_store(zz, gz + o);
      __builtin_nontemporal_store(hy, hn + o);
    }
  }
}

// ---------------------------------------------------------------------------
// FC: out[row, o] = H1cat[row, :] . fc_w[o, :] + fc_b[o]
// M=65536, N=256, K=1024. Workgroup = 16 rows (staged in LDS); 8 waves cover
// all 256 output columns (2 tiles each). Grid 4096.
// ---------------------------------------------------------------------------
__global__ __launch_bounds__(256)
void fc_gemm(const bf16* __restrict__ A,    // (B*T, 2H) bf16
             const bf16* __restrict__ Wfc,  // (OUT, 2H) bf16
             const float* __restrict__ bias,
             float* __restrict__ out) {
  constexpr int LDP = 1032;             // 1024 + 8 pad
  __shared__ bf16 sA[16 * LDP];
  const int tid  = threadIdx.x;
  const int lane = tid & 31;
  const int w    = tid >> 5;
  const size_t M0 = (size_t)blockIdx.x * 16;

  stage16(sA, LDP, A + M0 * (size_t)H2_, (size_t)H2_, H2_, 7, tid);
  stage_fence();

  v8f acc0, acc1;
  acc0 = 0.f; acc1 = 0.f;
  const int n0 = w * 32;
  const bf16* w0 = Wfc + (size_t)n0 * (size_t)H2_;
  const bf16* w1 = Wfc + (size_t)(n0 + 16) * (size_t)H2_;
  for (int k = 0; k < H2_; k += 32) {
    v16bf a = load_a(sA, LDP, lane, k);
    acc0 = wmma_bf16(a, load_b(w0, H2_, lane, k), acc0);
    acc1 = wmma_bf16(a, load_b(w1, H2_, lane, k), acc1);
  }

  const int half = lane >> 4;
  const int col  = lane & 15;
#pragma unroll
  for (int sdx = 0; sdx < 2; ++sdx) {
    const int j = n0 + sdx * 16 + col;
    const float bj = bias[j];
    const v8f acc = sdx ? acc1 : acc0;
#pragma unroll
    for (int v = 0; v < 8; ++v) {
      const size_t row = M0 + (size_t)(half * 8 + v);
      __builtin_nontemporal_store(acc[v] + bj, out + row * (size_t)OUT_ + j);
    }
  }
}

// ---------------------------------------------------------------------------
__global__ void f32_to_bf16(const float* __restrict__ src, bf16* __restrict__ dst,
                            size_t n) {
  size_t i = (size_t)blockIdx.x * blockDim.x + threadIdx.x;
  const size_t stride = (size_t)gridDim.x * blockDim.x;
  for (; i < n; i += stride) dst[i] = (bf16)src[i];
}

// ---------------------------------------------------------------------------
extern "C" void kernel_launch(void* const* d_in, const int* in_sizes, int n_in,
                              void* d_out, int out_size, void* d_ws, size_t ws_size,
                              hipStream_t stream) {
  const float* x        = (const float*)d_in[0];   // (B,T,IN)
  const float* w_ih_in  = (const float*)d_in[1];   // (3H,IN)
  const float* w_hh_in  = (const float*)d_in[2];   // (3H,H)
  const float* b_ih_in  = (const float*)d_in[3];
  const float* b_hh_in  = (const float*)d_in[4];
  const float* w_ih_mid = (const float*)d_in[5];   // (3H,H)
  const float* w_hh_mid = (const float*)d_in[6];   // (3H,H)
  const float* b_ih_mid = (const float*)d_in[7];
  const float* b_hh_mid = (const float*)d_in[8];
  const float* fc_w     = (const float*)d_in[9];   // (OUT,2H)
  const float* fc_b     = (const float*)d_in[10];

  // d_out partition (tuple order: outputs, gates_r, gates_z, hn)
  float* out = (float*)d_out;
  float* gr  = out + (size_t)B_ * T_ * OUT_;
  float* gz  = gr + (size_t)B_ * T_ * H2_;
  float* hn  = gz + (size_t)B_ * T_ * H2_;

  // workspace carve (bf16 scratch)
  size_t off = 0;
  char* ws = (char*)d_ws;
  auto carve = [&](size_t elems) -> bf16* {
    bf16* p = (bf16*)(ws + off);
    off += ((elems * sizeof(bf16)) + 255) & ~(size_t)255;
    return p;
  };
  bf16* Xbf    = carve((size_t)B_ * T_ * IN_);
  bf16* WihIn  = carve((size_t)3 * H_ * IN_);
  bf16* WhhIn  = carve((size_t)3 * H_ * H_);
  bf16* WihMid = carve((size_t)3 * H_ * H_);
  bf16* WhhMid = carve((size_t)3 * H_ * H_);
  bf16* Wfc    = carve((size_t)OUT_ * H2_);
  bf16* H0     = carve((size_t)2 * B_ * T_ * H_);   // per-dir layer-0 states
  bf16* H1     = carve((size_t)B_ * T_ * H2_);      // concat layer-1 states

  const int cb = 256;
  f32_to_bf16<<<2048, cb, 0, stream>>>(x, Xbf, (size_t)B_ * T_ * IN_);
  f32_to_bf16<<<512,  cb, 0, stream>>>(w_ih_in,  WihIn,  (size_t)3 * H_ * IN_);
  f32_to_bf16<<<512,  cb, 0, stream>>>(w_hh_in,  WhhIn,  (size_t)3 * H_ * H_);
  f32_to_bf16<<<512,  cb, 0, stream>>>(w_ih_mid, WihMid, (size_t)3 * H_ * H_);
  f32_to_bf16<<<512,  cb, 0, stream>>>(w_hh_mid, WhhMid, (size_t)3 * H_ * H_);
  f32_to_bf16<<<512,  cb, 0, stream>>>(fc_w,     Wfc,    (size_t)OUT_ * H2_);

  const dim3 sgrid(8, 4, 2);  // M-blocks(16 rows), N-blocks(128 cols), dirs
  // layer 0 (input = x, shared by both directions; per-dir state buffers)
  for (int s = 0; s < T_; ++s) {
    gru_step<<<sgrid, 256, 0, stream>>>(
        Xbf, IN_, (size_t)0,
        H0, (size_t)B_ * T_ * H_, H_, 0,
        WihIn, WhhIn, b_ih_in, b_hh_in,
        s, 0, nullptr, nullptr, nullptr);
  }
  // layer 1 (input = per-dir layer-0 states; state in concat buffer; outputs)
  for (int s = 0; s < T_; ++s) {
    gru_step<<<sgrid, 256, 0, stream>>>(
        H0, H_, (size_t)B_ * T_ * H_,
        H1, (size_t)0, H2_, H_,
        WihMid, WhhMid, b_ih_mid, b_hh_mid,
        s, 1, gr, gz, hn);
  }
  // final FC over concatenated states
  fc_gemm<<<4096, 256, 0, stream>>>(H1, Wfc, fc_b, out);
}
